// LSTM_14422500180491
// MI455X (gfx1250) — matrix operand order repaired and checked by reading, not compiled, not run on hardware
//
#include <hip/hip_runtime.h>
#include <hip/hip_bf16.h>
#include <math.h>

typedef __attribute__((ext_vector_type(16))) __bf16 v16bf;
typedef __attribute__((ext_vector_type(8)))  float  v8f;

#define BB 128
#define TT 1000
#define HH 512
#define GG 2048          // 4*H
#define NGROUP 8         // batch groups (independent recurrences)
#define WPG 8            // workgroups per group
#define MBR 16           // batch rows per group
#define JW 64            // hidden units per WG
#define NCOL 256         // gate columns per WG (4 gates * 64)
#define THREADS 256      // 8 waves

// dynamic LDS layout (bytes):
// wlds bf16[256][512] : 0      (262144)  resident W_hh slice, [col][k]
// hlds bf16[16][512]  : 262144 (16384)   staged h_{t-1} for this batch group
// glds f32 [16][256]  : 278528 (16384)   gate results
// clds f32 [16][64]   : 294912 (4096)    persistent cell state
// wih0 f32 [256]      : 299008
// wih1 f32 [256]      : 300032
// blds f32 [256]      : 301056
#define SMEM_BYTES 302080

union FragA { uint4 q[2]; v16bf v; };

__device__ __forceinline__ unsigned f2bf(float x) {
  unsigned u = __float_as_uint(x);
  return (u + 0x7FFFu + ((u >> 16) & 1u)) >> 16;
}
__device__ __forceinline__ float sigmoidf(float x) { return 1.0f / (1.0f + __expf(-x)); }

// Async DMA: 16 bytes global -> LDS, tracked by ASYNCcnt (no VGPR bounce).
__device__ __forceinline__ void async_copy16(const void* gptr, void* lptr) {
  unsigned lds = (unsigned)(size_t)lptr;               // addr[31:0] = LDS offset
  unsigned long long ga = (unsigned long long)(size_t)gptr;
  asm volatile("global_load_async_to_lds_b128 %0, %1, off"
               :: "v"(lds), "v"(ga) : "memory");
}

// Sync the 8 workgroups of one batch group. Real path: L2 atomics.
// Also emits the CDNA5 cluster barrier (NOP when not cluster-dispatched).
__device__ __forceinline__ void group_sync(unsigned* ctr, unsigned* gen) {
  __threadfence();
  asm volatile("s_wait_storecnt 0x0" ::: "memory");
  __syncthreads();
  if (threadIdx.x == 0) {
    unsigned arrive =
        __hip_atomic_fetch_add(ctr, 1u, __ATOMIC_RELEASE, __HIP_MEMORY_SCOPE_AGENT) + 1u;
    unsigned need = (arrive + WPG - 1u) / WPG;
    if ((arrive % WPG) == 0u)
      __hip_atomic_fetch_add(gen, 1u, __ATOMIC_RELEASE, __HIP_MEMORY_SCOPE_AGENT);
    while (__hip_atomic_load(gen, __ATOMIC_ACQUIRE, __HIP_MEMORY_SCOPE_AGENT) < need)
      __builtin_amdgcn_s_sleep(1);
  }
  if (threadIdx.x < 32) __builtin_amdgcn_s_cluster_barrier();
  __syncthreads();
  asm volatile("global_inv\n\ts_wait_loadcnt 0x0" ::: "memory");
}

__global__ __launch_bounds__(THREADS) void lstm_persistent(
    const float* __restrict__ x,     // [B, T, 2]
    const float* __restrict__ wih,   // [2, 4H]
    const float* __restrict__ whh,   // [H, 4H]
    const float* __restrict__ bias,  // [4H]
    float* __restrict__ out,         // h: B*T*H, then c: B*T*H
    unsigned short* __restrict__ hbuf,  // 2 * B*H bf16, double-buffered
    unsigned* __restrict__ barws)       // NGROUP * 64 u32
{
  extern __shared__ char smem[];
  unsigned short* wlds = (unsigned short*)smem;
  unsigned short* hlds = (unsigned short*)(smem + 262144);
  float* glds = (float*)(smem + 278528);
  float* clds = (float*)(smem + 294912);
  float* wih0 = (float*)(smem + 299008);
  float* wih1 = (float*)(smem + 300032);
  float* blds = (float*)(smem + 301056);

  const int tid   = threadIdx.x;
  const int grp   = blockIdx.x / WPG;   // batch group
  const int wgl   = blockIdx.x % WPG;   // wg within group
  const int bbase = grp * MBR;
  const int jbase = wgl * JW;

  unsigned* ctr = barws + grp * 64;
  unsigned* gen = ctr + 32;

  // ---- one-time preload: W_ih/bias slice, c0 = 0 ----
  {
    int qq = tid >> 6, jj = tid & 63;
    int colg = qq * HH + jbase + jj;
    wih0[tid] = wih[colg];
    wih1[tid] = wih[GG + colg];
    blds[tid] = bias[colg];
  }
  for (int i = tid; i < MBR * JW; i += THREADS) clds[i] = 0.0f;

  // ---- one-time: convert W_hh slice f32 -> bf16, transposed to [col][k] ----
  for (int i = tid; i < NCOL * HH; i += THREADS) {
    int cl = i & (NCOL - 1);
    int k  = i >> 8;
    int qq = cl >> 6, jj = cl & 63;
    int colg = qq * HH + jbase + jj;
    wlds[cl * HH + k] = (unsigned short)f2bf(whh[k * GG + colg]);
  }
  __syncthreads();

  const int wave = tid >> 5;          // 0..7
  const int gq   = wave >> 1;         // gate: 0=f 1=i 2=o 3=g
  const int hoff = (wave & 1) * 32;   // hidden sub-block within gate
  const int lane = tid & 31;
  const int ln   = lane & 15;
  const int lh   = lane >> 4;
  const int cb0  = gq * JW + hoff + ln;   // col_local, tile 0
  const int cb1  = cb0 + 16;              // col_local, tile 1

  for (int t = 0; t < TT; ++t) {
    const int rb = t & 1;        // read buffer (h_{t-1})
    const int wb = rb ^ 1;       // write buffer (h_t)

    if (t > 0) {
      // async-DMA h_{t-1} [16 x 512] bf16 for this batch group into LDS
      const unsigned short* hsrc = hbuf + rb * (BB * HH) + bbase * HH;
      for (int i = tid * 8; i < MBR * HH; i += THREADS * 8)
        async_copy16(hsrc + i, hlds + i);
      asm volatile("s_wait_asynccnt 0x0" ::: "memory");
      __syncthreads();

      // GEMM: [16 x 512] @ [512 x 32] for this wave (2 N-tiles of 16)
      v8f acc0 = {0.f, 0.f, 0.f, 0.f, 0.f, 0.f, 0.f, 0.f};
      v8f acc1 = acc0;
      #pragma unroll 4
      for (int kk = 0; kk < HH; kk += 32) {
        FragA a, b0, b1;
        int ha = ln * HH + kk + lh * 16;
        a.q[0] = *(const uint4*)(hlds + ha);
        a.q[1] = *(const uint4*)(hlds + ha + 8);
        int wa0 = cb0 * HH + kk + lh * 16;
        b0.q[0] = *(const uint4*)(wlds + wa0);
        b0.q[1] = *(const uint4*)(wlds + wa0 + 8);
        int wa1 = cb1 * HH + kk + lh * 16;
        b1.q[0] = *(const uint4*)(wlds + wa1);
        b1.q[1] = *(const uint4*)(wlds + wa1 + 8);
        acc0 = __builtin_amdgcn_wmma_f32_16x16x32_bf16(false, a.v, false, b0.v,
                                                       (short)0, acc0, false, false);
        acc1 = __builtin_amdgcn_wmma_f32_16x16x32_bf16(false, a.v, false, b1.v,
                                                       (short)0, acc1, false, false);
      }

      // add input contribution + bias, spill gates to LDS
      #pragma unroll
      for (int r = 0; r < 8; ++r) {
        int m = r + 8 * lh;            // C/D layout: VGPR r, lane half lh
        int b = bbase + m;
        const float2 xv = *(const float2*)(x + ((size_t)b * TT + t) * 2);
        glds[m * NCOL + cb0] = acc0[r] + xv.x * wih0[cb0] + xv.y * wih1[cb0] + blds[cb0];
        glds[m * NCOL + cb1] = acc1[r] + xv.x * wih0[cb1] + xv.y * wih1[cb1] + blds[cb1];
      }
      __syncthreads();
    }

    // cell update: 1024 (m, jj) elements, 4 per thread; keep h1/c1 in regs
    float h1v[4], c1v[4];
    #pragma unroll
    for (int r = 0; r < 4; ++r) {
      int e  = tid + r * THREADS;
      int m  = e >> 6;
      int jj = e & 63;
      int b  = bbase + m;
      float fg, ig, og, gg;
      if (t > 0) {
        fg = glds[m * NCOL + 0 * JW + jj];
        ig = glds[m * NCOL + 1 * JW + jj];
        og = glds[m * NCOL + 2 * JW + jj];
        gg = glds[m * NCOL + 3 * JW + jj];
      } else {  // h0 == 0: gates are input contribution only
        const float2 xv = *(const float2*)(x + ((size_t)b * TT + t) * 2);
        fg = xv.x * wih0[0 * JW + jj] + xv.y * wih1[0 * JW + jj] + blds[0 * JW + jj];
        ig = xv.x * wih0[1 * JW + jj] + xv.y * wih1[1 * JW + jj] + blds[1 * JW + jj];
        og = xv.x * wih0[2 * JW + jj] + xv.y * wih1[2 * JW + jj] + blds[2 * JW + jj];
        gg = xv.x * wih0[3 * JW + jj] + xv.y * wih1[3 * JW + jj] + blds[3 * JW + jj];
      }
      float c0v = clds[e];
      float c1  = sigmoidf(fg) * c0v + sigmoidf(ig) * tanhf(gg);
      float h1  = sigmoidf(og) * tanhf(c1);
      clds[e] = c1;
      hbuf[wb * (BB * HH) + b * HH + jbase + jj] = (unsigned short)f2bf(h1);
      h1v[r] = h1;
      c1v[r] = c1;
      if (t + 1 < TT)
        __builtin_prefetch(x + ((size_t)b * TT + t + 1) * 2, 0, 0);
    }

    group_sync(ctr, gen);   // only hbuf stores gate the recurrence

    // deferred f32 output stores: overlap next step's staging/GEMM
    #pragma unroll
    for (int r = 0; r < 4; ++r) {
      int e  = tid + r * THREADS;
      int m  = e >> 6;
      int jj = e & 63;
      int b  = bbase + m;
      size_t oidx = ((size_t)b * TT + t) * HH + jbase + jj;
      out[oidx] = h1v[r];
      out[(size_t)BB * TT * HH + oidx] = c1v[r];
    }
  }
}

extern "C" void kernel_launch(void* const* d_in, const int* in_sizes, int n_in,
                              void* d_out, int out_size, void* d_ws, size_t ws_size,
                              hipStream_t stream) {
  (void)in_sizes; (void)n_in; (void)out_size; (void)ws_size;
  const float* x    = (const float*)d_in[0];
  const float* wih  = (const float*)d_in[1];
  const float* whh  = (const float*)d_in[2];
  const float* bias = (const float*)d_in[3];
  float* out = (float*)d_out;

  unsigned short* hbuf = (unsigned short*)d_ws;                       // 2*B*H bf16 = 256 KB
  unsigned* barws = (unsigned*)((char*)d_ws + 2 * BB * HH * sizeof(unsigned short));

  hipMemsetAsync(barws, 0, NGROUP * 64 * sizeof(unsigned), stream);

  dim3 grid(NGROUP * WPG);   // 64 persistent WGs = 8 groups of 8
  dim3 block(THREADS);
  lstm_persistent<<<grid, block, SMEM_BYTES, stream>>>(x, wih, whh, bias, out, hbuf, barws);
}
